// CustomGRU_3435973837373
// MI455X (gfx1250) — matrix operand order repaired and checked by reading, not compiled, not run on hardware
//
#include <hip/hip_runtime.h>

typedef __bf16 bf16_t;
typedef __attribute__((ext_vector_type(16))) __bf16 v16bf;
typedef __attribute__((ext_vector_type(8)))  __bf16 v8bf;
typedef __attribute__((ext_vector_type(8)))  float  v8f;

#define B_   256
#define S_   512
#define I_   256
#define H_   512
#define KTOT (I_ + H_)      // 768: concatenated K = [x_t | h]
#define NROW 4              // row-stripes  (64 batch rows each)
#define NCOL 8              // col-slices   (64 H columns each)
#define NBLK (NROW * NCOL)  // 32 persistent blocks

// ---------------------------------------------------------------------------
// prep kernels
// ---------------------------------------------------------------------------
__global__ void cvt_f32_bf16(const float* __restrict__ src,
                             bf16_t* __restrict__ dst, int n) {
    int i = blockIdx.x * blockDim.x + threadIdx.x;
    int stride = gridDim.x * blockDim.x;
    for (; i < n; i += stride) dst[i] = (bf16_t)src[i];
}

__global__ void init_h(const float* __restrict__ h0,
                       float* __restrict__ h_f32,
                       bf16_t* __restrict__ h_bf,
                       unsigned* __restrict__ barcnt, int n) {
    if (blockIdx.x == 0 && threadIdx.x == 0) *barcnt = 0u;  // reset grid barrier
    int i = blockIdx.x * blockDim.x + threadIdx.x;
    int stride = gridDim.x * blockDim.x;
    for (; i < n; i += stride) {
        float v = h0[i];
        h_f32[i] = v;
        h_bf[i]  = (bf16_t)v;
    }
}

// ---------------------------------------------------------------------------
// split device-wide barrier (monotonic counter in L2, agent-scope fences).
// signal: all waves flush stores, then one lane bumps the counter.
// wait:   spin until all NBLK blocks signalled this round, then acquire.
// Independent WMMA work is placed between signal and wait to hide latency.
// ---------------------------------------------------------------------------
__device__ __forceinline__ void grid_signal(unsigned* cnt) {
    __builtin_amdgcn_fence(__ATOMIC_RELEASE, "agent");   // every wave flushes
    __syncthreads();
    if (threadIdx.x == 0)
        __hip_atomic_fetch_add(cnt, 1u, __ATOMIC_RELAXED, __HIP_MEMORY_SCOPE_AGENT);
}

__device__ __forceinline__ void grid_wait(unsigned* cnt, unsigned target) {
    if (threadIdx.x == 0) {
        while (__hip_atomic_load(cnt, __ATOMIC_RELAXED, __HIP_MEMORY_SCOPE_AGENT) < target)
            __builtin_amdgcn_s_sleep(1);
    }
    __syncthreads();
    __builtin_amdgcn_fence(__ATOMIC_ACQUIRE, "agent");
}

// ---------------------------------------------------------------------------
// persistent GRU scan kernel: one launch for all 512 timesteps.
// Block = 64 batch rows x 64 H cols; 4 waves tiled 2x2 (wave = 32x32).
// Weight slice (3 gates x 64 cols x 768 K bf16 = 288 KB) lives in LDS.
// ---------------------------------------------------------------------------
__global__ __launch_bounds__(128)
void gru_persist(const bf16_t* __restrict__ x_bf,
                 bf16_t* __restrict__ h_bf,
                 float*  __restrict__ h_f32,
                 bf16_t* __restrict__ rh_bf,
                 const bf16_t* __restrict__ Wir, const bf16_t* __restrict__ Whr,
                 const bf16_t* __restrict__ Wiz, const bf16_t* __restrict__ Whz,
                 const bf16_t* __restrict__ Wih, const bf16_t* __restrict__ Whh,
                 const float* __restrict__ br, const float* __restrict__ bz,
                 const float* __restrict__ bh,
                 float* __restrict__ out, float* __restrict__ h_last,
                 unsigned* __restrict__ barcnt)
{
    __shared__ __attribute__((aligned(32))) bf16_t wlds[3][64][KTOT]; // 288 KiB

    const int lane = threadIdx.x & 31;
    const int wave = threadIdx.x >> 5;
    const int wr   = wave >> 1;          // wave row group (0..1)
    const int wc   = wave & 1;           // wave col group (0..1)
    const int row0 = blockIdx.x * 64;    // batch-row stripe
    const int col0 = blockIdx.y * 64;    // H-column slice

    // ---- prologue: stage weight slice into LDS (once) ----------------------
    for (int idx = threadIdx.x; idx < 3 * 64 * (KTOT / 8); idx += 128) {
        const int g   = idx / (64 * (KTOT / 8));
        const int rem = idx % (64 * (KTOT / 8));
        const int c   = rem / (KTOT / 8);
        const int k8  = (rem % (KTOT / 8)) * 8;
        const bf16_t* wi = (g == 0) ? Wir : ((g == 1) ? Wiz : Wih);
        const bf16_t* wh = (g == 0) ? Whr : ((g == 1) ? Whz : Whh);
        const bf16_t* src = (k8 < I_) ? (wi + (col0 + c) * I_ + k8)
                                      : (wh + (col0 + c) * H_ + (k8 - I_));
        *(v8bf*)&wlds[g][c][k8] = *(const v8bf*)src;
    }
    __syncthreads();

    // WMMA operand addressing
    const int kAoff = (lane >> 4) * 16;   // A frag: 16 contiguous K per lane
    const int kBoff = (lane >> 4) * 16;   // B frag: 16 contiguous K per lane
    int mrow[2], nloc[2];
    #pragma unroll
    for (int mi = 0; mi < 2; ++mi) mrow[mi] = row0 + wr * 32 + mi * 16 + (lane & 15);
    #pragma unroll
    for (int ni = 0; ni < 2; ++ni) nloc[ni] = wc * 32 + ni * 16 + (lane & 15);
    // C/D mapping: lane L, VGPR v -> (M = v + 8*(L>>4), N = L&15)
    const int mD = row0 + wr * 32 + 8 * (lane >> 4);
    const int nD = col0 + wc * 32 + (lane & 15);

    unsigned sig = 0;   // completed signal rounds

    for (int s = 0; s < S_; ++s) {
        // ================= phase A: r and z gates ==========================
        v8f aR[2][2], aZ[2][2];
        #pragma unroll
        for (int mi = 0; mi < 2; ++mi)
            #pragma unroll
            for (int ni = 0; ni < 2; ++ni) {
                aR[mi][ni] = (v8f){0.f,0.f,0.f,0.f,0.f,0.f,0.f,0.f};
                aZ[mi][ni] = (v8f){0.f,0.f,0.f,0.f,0.f,0.f,0.f,0.f};
            }

        // --- A x-part (kc < I_): independent of h -> overlaps h publication
        for (int kc = 0; kc < I_; kc += 32) {
            union { v16bf v; v8bf h[2]; } af[2];
            #pragma unroll
            for (int mi = 0; mi < 2; ++mi) {
                const bf16_t* base = x_bf + ((size_t)mrow[mi] * S_ + s) * I_ + kc + kAoff;
                af[mi].h[0] = *(const v8bf*)base;
                af[mi].h[1] = *(const v8bf*)(base + 8);
            }
            #pragma unroll
            for (int ni = 0; ni < 2; ++ni) {
                v16bf bR = *(const v16bf*)&wlds[0][nloc[ni]][kc + kBoff];
                v16bf bZ = *(const v16bf*)&wlds[1][nloc[ni]][kc + kBoff];
                #pragma unroll
                for (int mi = 0; mi < 2; ++mi) {
                    aR[mi][ni] = __builtin_amdgcn_wmma_f32_16x16x32_bf16(
                        false, af[mi].v, false, bR, (short)0, aR[mi][ni], false, false);
                    aZ[mi][ni] = __builtin_amdgcn_wmma_f32_16x16x32_bf16(
                        false, af[mi].v, false, bZ, (short)0, aZ[mi][ni], false, false);
                }
            }
        }

        grid_wait(barcnt, sig * NBLK);   // h(s-1) published (no-op at s==0)

        // --- A h-part (recurrent K)
        for (int kc = 0; kc < H_; kc += 32) {
            union { v16bf v; v8bf h[2]; } af[2];
            #pragma unroll
            for (int mi = 0; mi < 2; ++mi) {
                const bf16_t* base = h_bf + mrow[mi] * H_ + kc + kAoff;
                af[mi].h[0] = *(const v8bf*)base;
                af[mi].h[1] = *(const v8bf*)(base + 8);
            }
            #pragma unroll
            for (int ni = 0; ni < 2; ++ni) {
                v16bf bR = *(const v16bf*)&wlds[0][nloc[ni]][I_ + kc + kBoff];
                v16bf bZ = *(const v16bf*)&wlds[1][nloc[ni]][I_ + kc + kBoff];
                #pragma unroll
                for (int mi = 0; mi < 2; ++mi) {
                    aR[mi][ni] = __builtin_amdgcn_wmma_f32_16x16x32_bf16(
                        false, af[mi].v, false, bR, (short)0, aR[mi][ni], false, false);
                    aZ[mi][ni] = __builtin_amdgcn_wmma_f32_16x16x32_bf16(
                        false, af[mi].v, false, bZ, (short)0, aZ[mi][ni], false, false);
                }
            }
        }

        // epilogue A: r = sig(.), z = sig(.) kept in regs, rh -> global bf16
        v8f zs[2][2];
        #pragma unroll
        for (int mi = 0; mi < 2; ++mi) {
            #pragma unroll
            for (int ni = 0; ni < 2; ++ni) {
                const int n = nD + ni * 16;
                const float brn = br[n];
                const float bzn = bz[n];
                #pragma unroll
                for (int v = 0; v < 8; ++v) {
                    const int m  = mD + mi * 16 + v;
                    const int ix = m * H_ + n;
                    float r = 1.0f / (1.0f + __expf(-(aR[mi][ni][v] + brn)));
                    float z = 1.0f / (1.0f + __expf(-(aZ[mi][ni][v] + bzn)));
                    zs[mi][ni][v] = z;
                    rh_bf[ix] = (bf16_t)(r * h_f32[ix]);
                }
            }
        }

        grid_signal(barcnt); ++sig;       // publish rh

        // ================= phase B: candidate + update =====================
        v8f aH[2][2];
        #pragma unroll
        for (int mi = 0; mi < 2; ++mi)
            #pragma unroll
            for (int ni = 0; ni < 2; ++ni)
                aH[mi][ni] = (v8f){0.f,0.f,0.f,0.f,0.f,0.f,0.f,0.f};

        // --- B x-part: independent of rh -> overlaps rh publication
        for (int kc = 0; kc < I_; kc += 32) {
            union { v16bf v; v8bf h[2]; } af[2];
            #pragma unroll
            for (int mi = 0; mi < 2; ++mi) {
                const bf16_t* base = x_bf + ((size_t)mrow[mi] * S_ + s) * I_ + kc + kAoff;
                af[mi].h[0] = *(const v8bf*)base;
                af[mi].h[1] = *(const v8bf*)(base + 8);
            }
            #pragma unroll
            for (int ni = 0; ni < 2; ++ni) {
                v16bf bH = *(const v16bf*)&wlds[2][nloc[ni]][kc + kBoff];
                #pragma unroll
                for (int mi = 0; mi < 2; ++mi) {
                    aH[mi][ni] = __builtin_amdgcn_wmma_f32_16x16x32_bf16(
                        false, af[mi].v, false, bH, (short)0, aH[mi][ni], false, false);
                }
            }
        }

        grid_wait(barcnt, sig * NBLK);    // rh published

        // --- B rh-part
        for (int kc = 0; kc < H_; kc += 32) {
            union { v16bf v; v8bf h[2]; } af[2];
            #pragma unroll
            for (int mi = 0; mi < 2; ++mi) {
                const bf16_t* base = rh_bf + mrow[mi] * H_ + kc + kAoff;
                af[mi].h[0] = *(const v8bf*)base;
                af[mi].h[1] = *(const v8bf*)(base + 8);
            }
            #pragma unroll
            for (int ni = 0; ni < 2; ++ni) {
                v16bf bH = *(const v16bf*)&wlds[2][nloc[ni]][I_ + kc + kBoff];
                #pragma unroll
                for (int mi = 0; mi < 2; ++mi) {
                    aH[mi][ni] = __builtin_amdgcn_wmma_f32_16x16x32_bf16(
                        false, af[mi].v, false, bH, (short)0, aH[mi][ni], false, false);
                }
            }
        }

        // epilogue B: h = (1-z)*h_prev + z*tanh(.)
        #pragma unroll
        for (int mi = 0; mi < 2; ++mi) {
            #pragma unroll
            for (int ni = 0; ni < 2; ++ni) {
                const int n = nD + ni * 16;
                const float bhn = bh[n];
                #pragma unroll
                for (int v = 0; v < 8; ++v) {
                    const int m  = mD + mi * 16 + v;
                    const int ix = m * H_ + n;
                    float ah = aH[mi][ni][v] + bhn;
                    float e  = __expf(-2.0f * ah);
                    float ht = (1.0f - e) / (1.0f + e);   // tanh
                    float z  = zs[mi][ni][v];
                    float hp = h_f32[ix];
                    float hn = hp + z * (ht - hp);
                    out[((size_t)m * S_ + s) * H_ + n] = hn;
                    h_f32[ix] = hn;
                    h_bf[ix]  = (bf16_t)hn;
                    if (s == S_ - 1) h_last[ix] = hn;
                }
            }
        }

        grid_signal(barcnt); ++sig;       // publish h (next step's A waits)
    }
}

// ---------------------------------------------------------------------------
extern "C" void kernel_launch(void* const* d_in, const int* in_sizes, int n_in,
                              void* d_out, int out_size, void* d_ws, size_t ws_size,
                              hipStream_t stream) {
    const float* x   = (const float*)d_in[0];
    const float* h0  = (const float*)d_in[1];
    const float* Wir = (const float*)d_in[2];
    const float* Wiz = (const float*)d_in[3];
    const float* Wih = (const float*)d_in[4];
    const float* Whr = (const float*)d_in[5];
    const float* Whz = (const float*)d_in[6];
    const float* Whh = (const float*)d_in[7];
    const float* br  = (const float*)d_in[8];
    const float* bz  = (const float*)d_in[9];
    const float* bh  = (const float*)d_in[10];

    float* out    = (float*)d_out;
    float* h_last = out + (size_t)B_ * S_ * H_;

    // workspace layout (all offsets 32B aligned)
    char* ws = (char*)d_ws;
    size_t off = 0;
    bf16_t* x_bf   = (bf16_t*)(ws + off); off += (size_t)B_ * S_ * I_ * 2;
    bf16_t* Wir_bf = (bf16_t*)(ws + off); off += (size_t)H_ * I_ * 2;
    bf16_t* Wiz_bf = (bf16_t*)(ws + off); off += (size_t)H_ * I_ * 2;
    bf16_t* Wih_bf = (bf16_t*)(ws + off); off += (size_t)H_ * I_ * 2;
    bf16_t* Whr_bf = (bf16_t*)(ws + off); off += (size_t)H_ * H_ * 2;
    bf16_t* Whz_bf = (bf16_t*)(ws + off); off += (size_t)H_ * H_ * 2;
    bf16_t* Whh_bf = (bf16_t*)(ws + off); off += (size_t)H_ * H_ * 2;
    float*  h_f32  = (float*)(ws + off);  off += (size_t)B_ * H_ * 4;
    bf16_t* h_bf   = (bf16_t*)(ws + off); off += (size_t)B_ * H_ * 2;
    bf16_t* rh_bf  = (bf16_t*)(ws + off); off += (size_t)B_ * H_ * 2;
    unsigned* barcnt = (unsigned*)(ws + off); off += 32;

    // one-time conversions + barrier reset
    {
        const int T = 256;
        cvt_f32_bf16<<<2048, T, 0, stream>>>(x,   x_bf,   B_ * S_ * I_);
        cvt_f32_bf16<<<256,  T, 0, stream>>>(Wir, Wir_bf, H_ * I_);
        cvt_f32_bf16<<<256,  T, 0, stream>>>(Wiz, Wiz_bf, H_ * I_);
        cvt_f32_bf16<<<256,  T, 0, stream>>>(Wih, Wih_bf, H_ * I_);
        cvt_f32_bf16<<<256,  T, 0, stream>>>(Whr, Whr_bf, H_ * H_);
        cvt_f32_bf16<<<256,  T, 0, stream>>>(Whz, Whz_bf, H_ * H_);
        cvt_f32_bf16<<<256,  T, 0, stream>>>(Whh, Whh_bf, H_ * H_);
        init_h<<<256, T, 0, stream>>>(h0, h_f32, h_bf, barcnt, B_ * H_);
    }

    // one persistent kernel for the whole scan
    dim3 grid(NROW, NCOL);   // 4 x 8 = 32 blocks, one per WGP
    gru_persist<<<grid, 128, 0, stream>>>(x_bf, h_bf, h_f32, rh_bf,
                                          Wir_bf, Whr_bf, Wiz_bf, Whz_bf,
                                          Wih_bf, Whh_bf,
                                          br, bz, bh, out, h_last, barcnt);
    (void)in_sizes; (void)n_in; (void)out_size; (void)ws_size;
}
